// SSIMcal_36704790512048
// MI455X (gfx1250) — matrix operand order repaired and checked by reading, not compiled, not run on hardware
//
#include <hip/hip_runtime.h>

// CDNA5 / gfx1250 fused SSIM kernel.
//
// Roofline: 201 MB compulsory HBM traffic @ 23.3 TB/s => ~8.6us floor; the
// 134 MB input set is L2-resident (192 MB), so halo re-reads never hit HBM.
// Arithmetic is offloaded to the matrix pipe as banded matmuls:
//   box(P) = V(ones band) * P(patch) * H(ones band)
// realized as v_wmma_f32_16x16x32_bf16 (K=32 covers the 26-tap reach, ones
// weights exact in bf16, f32 accumulation). 15 WMMAs per 16x16 output tile.
//
// All boundary handling lives in the two constant band operands:
//  - the 32-row / 32-col load window is SHIFTED into the image
//    (rowbase/colbase = clamp(origin-5, 0, 512-32)), so every load is valid
//    and needs no clamp or mask; the shift (dy/dx) is folded into the band
//    range test. Zero-padding taps simply have no band slot => exact.
//  - per-element addresses are base + compile-time constant row offsets that
//    fold into the global-load immediate offset field (zero address VALU).
//
// The vertical-pass C-layout result feeds the horizontal-pass B operand with
// no lane shuffles: matmul is invariant under a K permutation applied to both
// operands; in kmapA's K order the C->B register/lane-half
// structure puts column k in slot k, so both bands use kmapA unpermuted.

typedef __attribute__((ext_vector_type(16))) __bf16 v16bf;
typedef __attribute__((ext_vector_type(8)))  float  v8f;

#define IMG_H 512
#define IMG_W 512
#define PAD   5          // WIN=11 -> radius 5
#define TS    16
#define TILES_X 32
#define TILES_PER_IMG 1024
#define WAVES_PER_BLK 8
#define LDS_STRIDE 17    // 16+1 pad: conflict-free strided LDS transpose

// 16-bit A-operand K map (ISA 7.12.2, 16-bit A 16x32):
//   element e, lane-half h -> K = 2*((e>>1)&3) + (e&1) + 8*h + 16*(e>>3)
__device__ __forceinline__ int kmapA(int e, int h) {
  return ((e >> 1) & 3) * 2 + (e & 1) + 8 * h + 16 * (e >> 3);
}

// Contiguous-order K (k = 16h + e) -> logical column for the horizontal pass, induced by feeding
// the vertical C-layout result directly into the B operand:
//   bv[e] (e<8)  = tv[0][e]   : lo lanes col e,      hi lanes col e+8
//   bv[e] (e>=8) = tv[1][e-8] : lo lanes col e-8+16, hi lanes col e-8+24
// => col(k) = k with bits 3 and 4 swapped.
__device__ __forceinline__ int colmap(int k) {
  return (k & 7) | ((k & 8) << 1) | ((k & 16) >> 1);
}

__device__ __forceinline__ int clampi(int v, int lo, int hi) {
  int a = v < lo ? lo : v;
  return a > hi ? hi : a;
}

__global__ __launch_bounds__(WAVES_PER_BLK * 32)
void ssim_wmma_kernel(const float* __restrict__ img,
                      const float* __restrict__ ref,
                      const float* __restrict__ drange,
                      float* __restrict__ out,
                      int n_tiles) {
  __shared__ float lds_t[WAVES_PER_BLK][TS * LDS_STRIDE];

  const int lane = threadIdx.x & 31;
  const int l15  = lane & 15;
  const int hi   = lane >> 4;
  const int wave = threadIdx.x >> 5;

  int tile = blockIdx.x * WAVES_PER_BLK + wave;
  if (tile >= n_tiles) tile = 0;  // launch is exact; keep EXEC full for WMMA

  const int b  = tile >> 10;                 // 1024 tiles per image
  const int t  = tile & (TILES_PER_IMG - 1);
  const int ty = t >> 5;
  const int tx = t & (TILES_X - 1);
  const int y0 = ty * TS;
  const int x0 = tx * TS;

  // Shifted, always-in-image 32x32 load window (tile-uniform scalars).
  const int rowbase = clampi(y0 - PAD, 0, IMG_H - 32);
  const int colbase = clampi(x0 - PAD, 0, IMG_W - 32);
  const int dy = (y0 - PAD) - rowbase;       // in [-PAD, 11]
  const int dx = (x0 - PAD) - colbase;

  const size_t ibase = (size_t)b * IMG_H * IMG_W;
  const float* ib = img + ibase;
  const float* rb = ref + ibase;

  // ---- band constants (built once per wave, pure lane-local compares) ----
  // vbandB: vertical-pass B operand. Slot kb = kmapA(e,h) holds global row
  //         rowbase+kb = patch row kb-dy.  1 iff y+dy <= kb <= y+dy+10.
  // hbandA: horizontal-pass A operand, same kmapA K order; slot ca holds
  //         global col colbase+ca = patch col ca-dx. 1 iff x+dx<=ca<=x+dx+10.
  v16bf vbandB, hbandA;
  const int ylo = l15 + dy;                  // output y = l15 (B lanes = N)
  const int xlo = l15 + dx;                  // output x = l15 (A lanes = M)
#pragma unroll
  for (int e = 0; e < 16; ++e) {
    const int kb = kmapA(e, hi);
    const int ca = kmapA(e, hi);
    vbandB[e] = (__bf16)((ylo <= kb && kb <= ylo + 10) ? 1.0f : 0.0f);
    hbandA[e] = (__bf16)((xlo <= ca && ca <= xlo + 10) ? 1.0f : 0.0f);
  }

  const v8f zero8 = {0.f, 0.f, 0.f, 0.f, 0.f, 0.f, 0.f, 0.f};

  // ---- vertical pass:  tv[cg](M=col, N=y) = sum_row P[row][col]*band ----
  // A operand holds the patch with lanes = columns -> every global load
  // instruction fetches 2 full 64B row segments (perfectly coalesced), and
  // each element's row offset is a compile-time constant folded into the
  // load's immediate offset.
  v8f tvx[2], tvy[2], tvxx[2], tvyy[2], tvxy[2];
#pragma unroll
  for (int cg = 0; cg < 2; ++cg) {
    const size_t base = (size_t)(rowbase + 8 * hi) * IMG_W + colbase + l15 + 16 * cg;
    const float* pI = ib + base;
    const float* pR = rb + base;
    v16bf ax, ay, axx, ayy, axy;
#pragma unroll
    for (int e = 0; e < 16; ++e) {
      // A-layout row for this element (h folded into base): constant per e
      const int roff = (((e >> 1) & 3) * 2 + (e & 1) + 16 * (e >> 3)) * IMG_W;
      const float xv = pI[roff];
      const float yv = pR[roff];
      ax[e]  = (__bf16)xv;
      ay[e]  = (__bf16)yv;
      axx[e] = (__bf16)(xv * xv);   // moments formed in f32, rounded once
      ayy[e] = (__bf16)(yv * yv);
      axy[e] = (__bf16)(xv * yv);
    }
    tvx[cg]  = __builtin_amdgcn_wmma_f32_16x16x32_bf16(false, ax,  false, vbandB, (short)0, zero8, false, false);
    tvy[cg]  = __builtin_amdgcn_wmma_f32_16x16x32_bf16(false, ay,  false, vbandB, (short)0, zero8, false, false);
    tvxx[cg] = __builtin_amdgcn_wmma_f32_16x16x32_bf16(false, axx, false, vbandB, (short)0, zero8, false, false);
    tvyy[cg] = __builtin_amdgcn_wmma_f32_16x16x32_bf16(false, ayy, false, vbandB, (short)0, zero8, false, false);
    tvxy[cg] = __builtin_amdgcn_wmma_f32_16x16x32_bf16(false, axy, false, vbandB, (short)0, zero8, false, false);
  }

  // ---- horizontal pass: vertical C-layout feeds the B operand directly ----
  // No shuffles: in kmapA K order bv slot k is column k, as hbandA expects.
  auto hpass = [&](const v8f (&tv)[2]) -> v8f {
    v16bf bv;
#pragma unroll
    for (int e = 0; e < 8; ++e) {
      bv[e]     = (__bf16)tv[0][e];   // lo: cols 0-7,   hi: cols 8-15
      bv[e + 8] = (__bf16)tv[1][e];   // lo: cols 16-23, hi: cols 24-31
    }
    // outT(M = x_out, N = y) : transposed result, fixed up via LDS below.
    return __builtin_amdgcn_wmma_f32_16x16x32_bf16(false, hbandA, false, bv, (short)0, zero8, false, false);
  };

  const v8f sxT  = hpass(tvx);
  const v8f syT  = hpass(tvy);
  const v8f sxxT = hpass(tvxx);
  const v8f syyT = hpass(tvyy);
  const v8f sxyT = hpass(tvxy);

  // ---- per-pixel SSIM math (layout-agnostic) + LDS transpose ----
  const float inv_n    = 1.0f / 121.0f;     // box sums -> means
  const float cov_norm = 121.0f / 120.0f;
  const float dr = drange[b];
  const float C1 = (0.01f * dr) * (0.01f * dr);
  const float C2 = (0.03f * dr) * (0.03f * dr);

  float* ldsw = lds_t[wave];                // wave-private scratch
#pragma unroll
  for (int v = 0; v < 8; ++v) {
    const float ux  = sxT[v]  * inv_n;
    const float uy  = syT[v]  * inv_n;
    const float uxx = sxxT[v] * inv_n;
    const float uyy = syyT[v] * inv_n;
    const float uxy = sxyT[v] * inv_n;
    const float vx  = cov_norm * (uxx - ux * ux);
    const float vy  = cov_norm * (uyy - uy * uy);
    const float vxy = cov_norm * (uxy - ux * uy);
    const float A1 = 2.0f * ux * uy + C1;
    const float A2 = 2.0f * vxy + C2;
    const float B1 = ux * ux + uy * uy + C1;
    const float B2 = vx + vy + C2;
    const float S  = (A1 * A2) / (B1 * B2);
    // this lane holds (x_local = v + 8*hi, y_local = l15): store row-major
    ldsw[l15 * LDS_STRIDE + v + 8 * hi] = S;
  }

  // same-wave LDS exchange: DS ops are in-order per wave; just stop the
  // compiler from hoisting the reads above the writes and drain DScnt.
  asm volatile("s_wait_dscnt 0" ::: "memory");

  float* ob = out + ibase;
#pragma unroll
  for (int v = 0; v < 8; ++v) {
    const float s = ldsw[(v + 8 * hi) * LDS_STRIDE + l15];
    // coalesced: lanes 0-15 -> row y0+v, lanes 16-31 -> row y0+v+8
    ob[(size_t)(y0 + v + 8 * hi) * IMG_W + (x0 + l15)] = s;
  }
}

extern "C" void kernel_launch(void* const* d_in, const int* in_sizes, int n_in,
                              void* d_out, int out_size, void* d_ws, size_t ws_size,
                              hipStream_t stream) {
  const float* img = (const float*)d_in[0];
  const float* ref = (const float*)d_in[1];
  const float* dr  = (const float*)d_in[2];
  float* out = (float*)d_out;

  const int B = in_sizes[2];                       // batch = len(data_range)
  const int n_tiles = B * TILES_PER_IMG;           // 64 * 1024 = 65536
  const int blocks  = (n_tiles + WAVES_PER_BLK - 1) / WAVES_PER_BLK;  // 8192

  ssim_wmma_kernel<<<blocks, WAVES_PER_BLK * 32, 0, stream>>>(img, ref, dr, out, n_tiles);
}
